// DCNv2_58291296141527
// MI455X (gfx1250) — matrix operand order repaired and checked by reading, not compiled, not run on hardware
//
#include <hip/hip_runtime.h>
#include <math.h>

// ---- CDNA5 WMMA vector types -------------------------------------------
typedef __attribute__((ext_vector_type(16))) __bf16 v16bf;
typedef __attribute__((ext_vector_type(8)))  float  v8f;

#define CIN   64
#define COUT  128
#define HH    128
#define WW    128
#define KK9   9
#define KDIM  576            // CIN * 9
#define NPIX  65536          // 4 * 128 * 128
#define TN    16             // output pixels per dcn_gemm workgroup tile
#define NKS   18             // KDIM / 32 k-steps
#define OWB_ELEMS (2 * NKS * 32 * 16)   // padded offset-weight B-image

// ========================================================================
// Prep 1: offset-conv weights (27x576, pad->32) split to bf16 hi/lo and
// pre-swizzled into the WMMA B-matrix register image:
//   [ntile][ks][lane][e]  with  K = (lane>>4)*16 + e,  N = lane&15.
// ========================================================================
__global__ __launch_bounds__(256) void owsplit_kernel(
    const float* __restrict__ ow, __bf16* __restrict__ owbh,
    __bf16* __restrict__ owbl)
{
  int idx = blockIdx.x * 256 + threadIdx.x;
  if (idx < OWB_ELEMS) {
    int e    = idx & 15;
    int lane = (idx >> 4) & 31;
    int ks   = (idx >> 9) % NKS;
    int nt   = idx / (NKS * 512);
    int ch   = nt * 16 + (lane & 15);
    int k    = ks * 32 + (lane >> 4) * 16 + e;
    float v  = (ch < 27) ? ow[ch * KDIM + k] : 0.f;
    __bf16 hi = (__bf16)v;
    owbh[idx] = hi;
    owbl[idx] = (__bf16)(v - (float)hi);
  }
}

// ========================================================================
// Prep 2: main conv weights fp32 -> bf16 (hi, lo) split (row-major).
// ========================================================================
__global__ __launch_bounds__(256) void wsplit_kernel(
    const float* __restrict__ w, __bf16* __restrict__ whi,
    __bf16* __restrict__ wlo)
{
  int i = blockIdx.x * 256 + threadIdx.x;
  if (i < COUT * KDIM) {
    float v = w[i];
    __bf16 hi = (__bf16)v;
    whi[i] = hi;
    wlo[i] = (__bf16)(v - (float)hi);
  }
}

// ========================================================================
// Kernel A (WMMA): offset conv as implicit GEMM.
//   A = im2col(x) built per-wave directly in the 16-bit A register layout
//       (contiguous K order: lanes 0-15 K 0-15, lanes 16-31 K 16-31),
//   B = padded offset weights (pre-swizzled, 2 N-tiles of 16 channels),
//   D = 16 pixels x 32 channels, fp32 accum, split-bf16 3-term product.
// Grid: NPIX/128 blocks x 256 threads; wave wv owns pixels wv*16..+15.
// ========================================================================
__global__ __launch_bounds__(256) void offconv_wmma_kernel(
    const float* __restrict__ x,
    const __bf16* __restrict__ owbh, const __bf16* __restrict__ owbl,
    const float* __restrict__ ob, float* __restrict__ off_raw)
{
  const int tid  = threadIdx.x;
  const int lane = tid & 31;
  const int wv   = tid >> 5;
  const int half = lane >> 4;
  const int n    = blockIdx.x * 128 + wv * 16 + (lane & 15); // sampled pixel
  const int b = n >> 14, s = n & 16383, h = s >> 7, w = s & 127;
  const float* xb = x + ((b * CIN) << 14);

  v8f acc0 = {0.f, 0.f, 0.f, 0.f, 0.f, 0.f, 0.f, 0.f};
  v8f acc1 = {0.f, 0.f, 0.f, 0.f, 0.f, 0.f, 0.f, 0.f};

  for (int ks = 0; ks < NKS; ++ks) {
    const int k0 = ks * 32;
    v16bf ah, al;
    #pragma unroll
    for (int e = 0; e < 16; ++e) {
      int K = half * 16 + e;
      int k = k0 + K;
      int c = k / KK9, p = k - c * KK9;
      int yy = h - 1 + p / 3, xx = w - 1 + p % 3;
      bool ok = (yy >= 0) & (yy < HH) & (xx >= 0) & (xx < WW);
      float v = ok ? xb[(c << 14) + yy * WW + xx] : 0.f;   // coalesced/lane
      __bf16 hi = (__bf16)v;
      ah[e] = hi;
      al[e] = (__bf16)(v - (float)hi);
    }
    const int boff = (ks * 32 + lane) * 16;
    v16bf bh0 = *(const v16bf*)(owbh + boff);
    v16bf bl0 = *(const v16bf*)(owbl + boff);
    v16bf bh1 = *(const v16bf*)(owbh + (NKS * 512) + boff);
    v16bf bl1 = *(const v16bf*)(owbl + (NKS * 512) + boff);

    acc0 = __builtin_amdgcn_wmma_f32_16x16x32_bf16(false, ah, false, bh0,
                                                   (short)0, acc0, false, false);
    acc0 = __builtin_amdgcn_wmma_f32_16x16x32_bf16(false, ah, false, bl0,
                                                   (short)0, acc0, false, false);
    acc0 = __builtin_amdgcn_wmma_f32_16x16x32_bf16(false, al, false, bh0,
                                                   (short)0, acc0, false, false);
    acc1 = __builtin_amdgcn_wmma_f32_16x16x32_bf16(false, ah, false, bh1,
                                                   (short)0, acc1, false, false);
    acc1 = __builtin_amdgcn_wmma_f32_16x16x32_bf16(false, ah, false, bl1,
                                                   (short)0, acc1, false, false);
    acc1 = __builtin_amdgcn_wmma_f32_16x16x32_bf16(false, al, false, bh1,
                                                   (short)0, acc1, false, false);
  }

  // D layout: VGPR g -> pixel row g + 8*half, channel = lane&15 (+16 for nt1)
  #pragma unroll
  for (int g = 0; g < 8; ++g) {
    int np  = blockIdx.x * 128 + wv * 16 + g + half * 8;
    int ss  = np & 16383;
    int ch0 = lane & 15;
    off_raw[((b * 27 + ch0) << 14) + ss] = acc0[g] + ob[ch0];
    int ch1 = ch0 + 16;
    if (ch1 < 27)
      off_raw[((b * 27 + ch1) << 14) + ss] = acc1[g] + ob[ch1];
  }
}

// ========================================================================
// Kernel B: fused deformable im2col + split-bf16 WMMA GEMM (unchanged).
// ========================================================================
__global__ __launch_bounds__(256) void dcn_gemm_kernel(
    const float* __restrict__ x, const float* __restrict__ off_raw,
    const __bf16* __restrict__ whi, const __bf16* __restrict__ wlo,
    const float* __restrict__ bias, float* __restrict__ out)
{
  __shared__ __bf16 colh[NKS * 32 * 16];   // 18 KB, B register image
  __shared__ __bf16 coll[NKS * 32 * 16];   // 18 KB
  __shared__ int    sidx[TN * KK9 * 4];    // 4 bilinear corner indices
  __shared__ float  swt [TN * KK9 * 4];    // 4 corner weights (mask-folded)

  const int tid = threadIdx.x;
  const int n0  = blockIdx.x * TN;
  const int b   = n0 >> 14;
  const int s0  = n0 & 16383;

  // ---- Phase 1: bilinear sampling geometry (16 px x 9 taps = 144 items)
  if (tid < TN * KK9) {
    int j = tid / KK9, p = tid - j * KK9;
    int s = s0 + j, h = s >> 7, w = s & 127;
    int ky = p / 3, kx = p - ky * 3;
    int ob14 = (b * 27) << 14;
    float offy = off_raw[ob14 + ((2 * p) << 14) + s];
    float offx = off_raw[ob14 + ((2 * p + 1) << 14) + s];
    float mraw = off_raw[ob14 + ((18 + p) << 14) + s];
    float mval = 1.f / (1.f + expf(-mraw));          // sigmoid(mask)
    float py = (float)(h - 1 + ky) + offy;
    float px = (float)(w - 1 + kx) + offx;
    float y0f = floorf(py), x0f = floorf(px);
    float ly = py - y0f, lx = px - x0f;
    int y0 = (int)y0f, x0 = (int)x0f;
    int cy[2] = {y0, y0 + 1};
    int cx[2] = {x0, x0 + 1};
    float wy[2] = {1.f - ly, ly};
    float wx[2] = {1.f - lx, lx};
    #pragma unroll
    for (int a = 0; a < 2; ++a) {
      #pragma unroll
      for (int d = 0; d < 2; ++d) {
        int yy = cy[a], xx = cx[d];
        bool ok = (yy >= 0) & (yy < HH) & (xx >= 0) & (xx < WW);
        int yyc = yy < 0 ? 0 : (yy > HH - 1 ? HH - 1 : yy);
        int xxc = xx < 0 ? 0 : (xx > WW - 1 ? WW - 1 : xx);
        sidx[tid * 4 + a * 2 + d] = yyc * WW + xxc;
        swt [tid * 4 + a * 2 + d] = ok ? (mval * wy[a] * wx[d]) : 0.f;
      }
    }
  }
  __syncthreads();

  // ---- Phase 2: build 576x16 column tile, split to bf16 hi/lo in LDS
  {
    int j  = tid & 15;        // pixel in tile
    int kb = tid >> 4;        // k residue 0..15
    #pragma unroll 4
    for (int i = 0; i < (KDIM * TN) / 256; ++i) {   // 36 values/thread
      int k = kb + i * 16;                          // 0..575
      int c = k / KK9, p = k - c * KK9;
      int it4 = (j * KK9 + p) * 4;
      const float* xp = x + ((b * CIN + c) << 14);
      float v = swt[it4 + 0] * xp[sidx[it4 + 0]]
              + swt[it4 + 1] * xp[sidx[it4 + 1]]
              + swt[it4 + 2] * xp[sidx[it4 + 2]]
              + swt[it4 + 3] * xp[sidx[it4 + 3]];
      __bf16 hi = (__bf16)v;
      __bf16 lo = (__bf16)(v - (float)hi);
      int ks = k >> 5, kr = k & 31;
      int lane = ((kr & 8) << 1) | j;
      int e    = (kr & 7) | ((kr & 16) >> 1);
      int pos  = (ks * 32 + lane) * 16 + e;
      colh[pos] = hi;
      coll[pos] = lo;
    }
  }
  __syncthreads();

  // ---- Phase 3: WMMA GEMM, one 16x16 M-tile per wave
  const int lane = tid & 31;
  const int wv   = tid >> 5;          // wave id 0..7 -> Cout rows wv*16..+15
  const int half = lane >> 4;
  const int mrow = lane & 15;
  const int grow = wv * 16 + mrow;    // A row this lane carries
  const __bf16* wh_row = whi + grow * KDIM;
  const __bf16* wl_row = wlo + grow * KDIM;

  v8f acc = {0.f, 0.f, 0.f, 0.f, 0.f, 0.f, 0.f, 0.f};

  for (int ks = 0; ks < NKS; ++ks) {
    const int k0 = ks * 32;
    union { uint4 q[2]; v16bf v; } ah, al;
    ah.q[0] = *(const uint4*)(wh_row + k0 + half * 8);
    ah.q[1] = *(const uint4*)(wh_row + k0 + 16 + half * 8);
    al.q[0] = *(const uint4*)(wl_row + k0 + half * 8);
    al.q[1] = *(const uint4*)(wl_row + k0 + 16 + half * 8);
    __builtin_prefetch((const void*)(wh_row + k0 + 32), 0, 1);

    v16bf bh = *(const v16bf*)&colh[(ks * 32 + lane) * 16];
    v16bf bl = *(const v16bf*)&coll[(ks * 32 + lane) * 16];

    acc = __builtin_amdgcn_wmma_f32_16x16x32_bf16(false, ah.v, false, bh,
                                                  (short)0, acc, false, false);
    acc = __builtin_amdgcn_wmma_f32_16x16x32_bf16(false, ah.v, false, bl,
                                                  (short)0, acc, false, false);
    acc = __builtin_amdgcn_wmma_f32_16x16x32_bf16(false, al.v, false, bh,
                                                  (short)0, acc, false, false);
  }

  // ---- Epilogue: C/D layout: VGPR g -> M = g + 8*half, N = lane&15
  #pragma unroll
  for (int g = 0; g < 8; ++g) {
    int o  = wv * 16 + g + half * 8;
    int nc = n0 + (lane & 15);
    out[((b * COUT + o) << 14) + (nc & 16383)] = acc[g] + bias[o];
  }
}

// ========================================================================
extern "C" void kernel_launch(void* const* d_in, const int* in_sizes, int n_in,
                              void* d_out, int out_size, void* d_ws, size_t ws_size,
                              hipStream_t stream) {
  const float* x   = (const float*)d_in[0];
  const float* ow  = (const float*)d_in[1];
  const float* ob  = (const float*)d_in[2];
  const float* wgt = (const float*)d_in[3];
  const float* bs  = (const float*)d_in[4];
  float* out = (float*)d_out;

  char* ws = (char*)d_ws;
  float*  off_raw = (float*)ws;                            // 7,077,888 B
  __bf16* whi  = (__bf16*)(ws + 7077888);                  //   147,456 B
  __bf16* wlo  = (__bf16*)(ws + 7077888 + 147456);         //   147,456 B
  __bf16* owbh = (__bf16*)(ws + 7372800);                  //    36,864 B
  __bf16* owbl = (__bf16*)(ws + 7372800 + 36864);          //    36,864 B

  owsplit_kernel<<<(OWB_ELEMS + 255) / 256, 256, 0, stream>>>(ow, owbh, owbl);
  wsplit_kernel<<<(COUT * KDIM + 255) / 256, 256, 0, stream>>>(wgt, whi, wlo);
  offconv_wmma_kernel<<<NPIX / 128, 256, 0, stream>>>(x, owbh, owbl, ob, off_raw);
  dcn_gemm_kernel<<<NPIX / TN, 256, 0, stream>>>(x, off_raw, whi, wlo, bs, out);
}
